// Attention_74680891343036
// MI455X (gfx1250) — matrix operand order repaired and checked by reading, not compiled, not run on hardware
//
#include <hip/hip_runtime.h>
#include <hip/hip_bf16.h>
#include <cstdint>

#define TSEQ   2048
#define DMODEL 3584
#define NHEADS 28
#define KVHEADS 4
#define HDIM   128
#define WQKV   4608   // 3584 Q + 512 K + 512 V
#define KCOL   3584
#define VCOL   4096
#define GROUP  7
#define SB     64     // attention S-block

typedef _Float16 half8 __attribute__((ext_vector_type(8)));
typedef _Float16 v16h  __attribute__((ext_vector_type(16)));
typedef float    v8f   __attribute__((ext_vector_type(8)));
typedef float    f4    __attribute__((ext_vector_type(4)));
typedef int      v4i   __attribute__((ext_vector_type(4)));

union Frag16 { v16h v; half8 h[2]; };

// ---------------------------------------------------------------------------
// CDNA5 async global->LDS copy path (ASYNCcnt-tracked), with sync fallback.
// ---------------------------------------------------------------------------
#if __has_builtin(__builtin_amdgcn_global_load_async_to_lds_b128)
#define USE_ASYNC_LDS 1
#else
#define USE_ASYNC_LDS 0
#endif

#if USE_ASYNC_LDS
typedef __attribute__((address_space(1))) v4i* as1_v4i;
typedef __attribute__((address_space(3))) v4i* as3_v4i;
#if __has_builtin(__builtin_amdgcn_s_wait_asynccnt)
#define WAIT_ASYNC(N) __builtin_amdgcn_s_wait_asynccnt(N)
#else
#define WAIT_ASYNC(N) asm volatile("s_wait_asynccnt %0" ::"i"(N) : "memory")
#endif
#else
#define WAIT_ASYNC(N) ((void)0)
#endif

// copy 16 bytes global -> LDS (per lane)
__device__ __forceinline__ void cp16B(void* lds_dst, const void* gsrc) {
#if USE_ASYNC_LDS
  __builtin_amdgcn_global_load_async_to_lds_b128(
      (as1_v4i)gsrc, (as3_v4i)lds_dst, 0, 0);
#else
  *(half8*)lds_dst = *(const half8*)gsrc;
#endif
}

__device__ __forceinline__ v8f zero8() {
  v8f z = {0.f,0.f,0.f,0.f,0.f,0.f,0.f,0.f};
  return z;
}

__device__ __forceinline__ v8f wmma_f16(const Frag16& a, const Frag16& b, v8f c) {
  return __builtin_amdgcn_wmma_f32_16x16x32_f16(
      false, a.v, false, b.v, (short)0, c, false, false);
}

// ---------------------------------------------------------------------------
// x: f32 -> f16 (so GEMM A-tiles can use verbatim async copies)
// ---------------------------------------------------------------------------
__global__ void __launch_bounds__(256) cvt_f32_f16_kernel(
    const float* __restrict__ src, _Float16* __restrict__ dst, int n8) {
  int id = blockIdx.x * 256 + threadIdx.x;
  if (id >= n8) return;
  f4 a0 = *(const f4*)(src + (size_t)id * 8);
  f4 a1 = *(const f4*)(src + (size_t)id * 8 + 4);
  half8 h;
  #pragma unroll
  for (int q = 0; q < 4; q++) { h[q] = (_Float16)a0[q]; h[4+q] = (_Float16)a1[q]; }
  *(half8*)(dst + (size_t)id * 8) = h;
}

// ---------------------------------------------------------------------------
// Weight transpose + f32->f16 convert: dst[n][k] = (f16) src[k][n]
// ---------------------------------------------------------------------------
__global__ void __launch_bounds__(256) transpose_cvt_kernel(
    const float* __restrict__ src, _Float16* __restrict__ dst, int K, int W) {
  __shared__ float tile[32][33];
  const int tx = threadIdx.x, ty = threadIdx.y;          // block (32,8)
  const int x  = blockIdx.x * 32 + tx;                   // src col (n)
  const int y0 = blockIdx.y * 32;                        // src row base (k)
  #pragma unroll
  for (int j = 0; j < 4; j++) {
    int y = y0 + ty + j * 8;
    if (x < W && y < K) tile[ty + j * 8][tx] = src[(size_t)y * W + x];
  }
  __syncthreads();
  const int k  = y0 + tx;
  const int n0 = blockIdx.x * 32;
  #pragma unroll
  for (int j = 0; j < 4; j++) {
    int n = n0 + ty + j * 8;
    if (k < K && n < W) dst[(size_t)n * K + k] = (_Float16)tile[tx][ty + j * 8];
  }
}

// ---------------------------------------------------------------------------
// RoPE in-place on f16 qkv buffer (Q heads 0..27, K heads 28..31)
// ---------------------------------------------------------------------------
__global__ void __launch_bounds__(256) rope_kernel(
    _Float16* __restrict__ qkv, const float* __restrict__ sinp,
    const float* __restrict__ cosp) {
  int id   = blockIdx.x * 256 + threadIdx.x;   // T * 32 heads * 64 pairs
  int i    = id & 63;
  int head = (id >> 6) & 31;
  int t    = id >> 11;
  if (t >= TSEQ) return;
  int col = (head < NHEADS) ? head * HDIM : KCOL + (head - NHEADS) * HDIM;
  _Float16* p = qkv + (size_t)t * WQKV + col;
  float s  = sinp[t * 64 + i];
  float c  = cosp[t * 64 + i];
  float x1 = (float)p[i];
  float x2 = (float)p[64 + i];
  p[i]      = (_Float16)(x1 * c - x2 * s);
  p[64 + i] = (_Float16)(x2 * c + x1 * s);
}

// ---------------------------------------------------------------------------
// V transpose: vt[kh*128 + h][t] = qkv[t][VCOL + kh*128 + h]
// ---------------------------------------------------------------------------
__global__ void __launch_bounds__(256) vtrans_kernel(
    const _Float16* __restrict__ qkv, _Float16* __restrict__ vt) {
  int id = blockIdx.x * 256 + threadIdx.x;    // KVHEADS*HDIM*TSEQ, t fastest
  int t  = id & (TSEQ - 1);
  int hh = id >> 11;                          // 0..511
  vt[(size_t)hh * TSEQ + t] = qkv[(size_t)t * WQKV + VCOL + hh];
}

// ---------------------------------------------------------------------------
// WMMA GEMM: C[M x N] = A[M x K] * Bt[N x K]^T (+ bias), 128x128 tile / WG.
// 8 waves: 4 in M (32 rows) x 2 in N (64 cols) -> 8 WMMA per 32-k step.
// Double-buffered LDS; async tile prefetch overlaps WMMA compute.
// ---------------------------------------------------------------------------
template <bool ADD_BIAS, typename OT>
__global__ void __launch_bounds__(256) gemm_wmma_kernel(
    const _Float16* __restrict__ A, const _Float16* __restrict__ Bt,
    const float* __restrict__ bias, OT* __restrict__ C,
    int M, int K, int ldC) {
  __shared__ _Float16 As[2][128][40];   // [buf][m][k]
  __shared__ _Float16 Bs[2][128][40];   // [buf][n][k]
  const int tid  = threadIdx.x;
  const int lane = tid & 31, wave = tid >> 5;
  const int khalf = lane >> 4, lan = lane & 15;
  const int m0 = blockIdx.y * 128, n0 = blockIdx.x * 128;
  const int wm = (wave >> 1) * 32, wn = (wave & 1) * 64;
  const int ldr = tid & 127, seg = tid >> 7;   // cooperative load row/segment

  const _Float16* aSrc = A  + (size_t)(m0 + ldr) * K + seg * 16;
  const _Float16* bSrc = Bt + (size_t)(n0 + ldr) * K + seg * 16;

  auto issueTile = [&](int kt, int buf) {
    cp16B(&As[buf][ldr][seg * 16],     aSrc + kt);
    cp16B(&As[buf][ldr][seg * 16 + 8], aSrc + kt + 8);
    cp16B(&Bs[buf][ldr][seg * 16],     bSrc + kt);
    cp16B(&Bs[buf][ldr][seg * 16 + 8], bSrc + kt + 8);
  };

  v8f acc[2][4];
  #pragma unroll
  for (int mf = 0; mf < 2; mf++)
    #pragma unroll
    for (int nf = 0; nf < 4; nf++) acc[mf][nf] = zero8();

  const int ntiles = K / 32;
  issueTile(0, 0);                          // prefetch tile 0
  for (int it = 0; it < ntiles; it++) {
    const int cur = it & 1;
    __syncthreads();                        // all waves done reading buf cur^1
    if (it + 1 < ntiles) {
      issueTile((it + 1) * 32, cur ^ 1);    // prefetch next tile
      WAIT_ASYNC(4);                        // own 4 oldest (tile it) landed
    } else {
      WAIT_ASYNC(0);
    }
    __syncthreads();                        // tile it visible to all waves

    Frag16 af[2], bf[4];
    #pragma unroll
    for (int mf = 0; mf < 2; mf++) {
      af[mf].h[0] = *(const half8*)&As[cur][wm + mf * 16 + lan][khalf * 8];
      af[mf].h[1] = *(const half8*)&As[cur][wm + mf * 16 + lan][16 + khalf * 8];
    }
    #pragma unroll
    for (int nf = 0; nf < 4; nf++) {
      bf[nf].h[0] = *(const half8*)&Bs[cur][wn + nf * 16 + lan][khalf * 8];
      bf[nf].h[1] = *(const half8*)&Bs[cur][wn + nf * 16 + lan][16 + khalf * 8];
    }
    #pragma unroll
    for (int nf = 0; nf < 4; nf++)
      #pragma unroll
      for (int mf = 0; mf < 2; mf++)
        acc[mf][nf] = wmma_f16(af[mf], bf[nf], acc[mf][nf]);
  }

  #pragma unroll
  for (int mf = 0; mf < 2; mf++)
    #pragma unroll
    for (int nf = 0; nf < 4; nf++) {
      int col = n0 + wn + nf * 16 + lan;
      float b = ADD_BIAS ? bias[col] : 0.0f;
      #pragma unroll
      for (int j = 0; j < 8; j++) {
        int row = m0 + wm + mf * 16 + ((lane & 16) ? 8 + j : j);
        C[(size_t)row * ldC + col] = (OT)(acc[mf][nf][j] + b);
      }
    }
}

// ---------------------------------------------------------------------------
// Flash attention (causal, GQA). grid = (T/128 q-tiles, NHEADS).
// 8 waves x 16 q-rows; S in 64-wide blocks, online softmax, async tile fills.
// ---------------------------------------------------------------------------
__global__ void __launch_bounds__(256) flash_attn_kernel(
    const _Float16* __restrict__ qkv, const _Float16* __restrict__ vt,
    _Float16* __restrict__ ctx) {
  __shared__ _Float16 Ks[SB][136];      // [s][h]   17408 B
  __shared__ _Float16 Vts[HDIM][72];    // [h][s]   18432 B
  __shared__ _Float16 Ps[128][72];      // [q][s]   18432 B  (total 54272 B)
  const int tid  = threadIdx.x;
  const int lane = tid & 31, wave = tid >> 5;
  const int khalf = lane >> 4, lan = lane & 15;
  const int qtile = blockIdx.x, head = blockIdx.y;
  const int kh = head / GROUP;
  const int q0 = qtile * 128;
  const float scale = 0.08838834764831845f;   // 1/sqrt(128)

  // Q A-fragments for this wave's 16 rows, loaded once (4 k-steps over H=128)
  Frag16 qa[4];
  {
    const _Float16* qp = qkv + (size_t)(q0 + wave * 16 + lan) * WQKV + head * HDIM;
    #pragma unroll
    for (int kk = 0; kk < 4; kk++) {
      qa[kk].h[0] = *(const half8*)(qp + kk * 32 + khalf * 8);
      qa[kk].h[1] = *(const half8*)(qp + kk * 32 + 16 + khalf * 8);
    }
  }

  // cooperative tile-fill addresses
  const int kr = tid >> 2, ksg = tid & 3;     // K tile: row 0..63, 4x64B segs
  const int vh = tid >> 1, vsg = tid & 1;     // V tile: h 0..127, 2x64B segs
  const _Float16* kSrc = qkv + (size_t)kr * WQKV + KCOL + kh * HDIM + ksg * 32;
  const _Float16* vSrc = vt + (size_t)(kh * HDIM + vh) * TSEQ + vsg * 32;

  v8f o[8];
  float mrow[8], lrow[8];
  #pragma unroll
  for (int f = 0; f < 8; f++) o[f] = zero8();
  #pragma unroll
  for (int j = 0; j < 8; j++) { mrow[j] = -3.0e38f; lrow[j] = 0.f; }

  const int nblocks = (qtile + 1) * (128 / SB);
  for (int sb = 0; sb < nblocks; sb++) {
    const int s0 = sb * SB;
    __syncthreads();                      // previous block's compute done
    // K tile: SB x 128 halves [s][h] — 4 concurrent async 16B copies
    {
      const _Float16* src = kSrc + (size_t)s0 * WQKV;
      _Float16* dst = &Ks[kr][ksg * 32];
      cp16B(dst,      src);
      cp16B(dst + 8,  src + 8);
      cp16B(dst + 16, src + 16);
      cp16B(dst + 24, src + 24);
    }
    // Vt tile: 128 x SB halves [h][s] — 4 concurrent async 16B copies
    {
      const _Float16* src = vSrc + s0;
      _Float16* dst = &Vts[vh][vsg * 32];
      cp16B(dst,      src);
      cp16B(dst + 8,  src + 8);
      cp16B(dst + 16, src + 16);
      cp16B(dst + 24, src + 24);
    }
    WAIT_ASYNC(0);
    __syncthreads();

    // ---- Q K^T ----
    v8f sc[SB / 16];
    #pragma unroll
    for (int sf = 0; sf < SB / 16; sf++) {
      Frag16 bf[4];
      #pragma unroll
      for (int kk = 0; kk < 4; kk++) {
        bf[kk].h[0] = *(const half8*)&Ks[sf * 16 + lan][kk * 32 + khalf * 8];
        bf[kk].h[1] = *(const half8*)&Ks[sf * 16 + lan][kk * 32 + 16 + khalf * 8];
      }
      v8f z = zero8();
      #pragma unroll
      for (int kk = 0; kk < 4; kk++) z = wmma_f16(qa[kk], bf[kk], z);
      sc[sf] = z;
    }
    // scale + causal mask
    #pragma unroll
    for (int sf = 0; sf < SB / 16; sf++) {
      int sabs = s0 + sf * 16 + lan;
      #pragma unroll
      for (int j = 0; j < 8; j++) {
        int rabs = q0 + wave * 16 + ((lane & 16) ? 8 + j : j);
        float v = sc[sf][j] * scale;
        sc[sf][j] = (sabs <= rabs) ? v : -2.3819763e38f;
      }
    }
    // ---- online softmax (rows live across 16-lane halves) ----
    #pragma unroll
    for (int j = 0; j < 8; j++) {
      float mx = mrow[j];
      #pragma unroll
      for (int sf = 0; sf < SB / 16; sf++) mx = fmaxf(mx, sc[sf][j]);
      mx = fmaxf(mx, __shfl_xor(mx, 1, 16));
      mx = fmaxf(mx, __shfl_xor(mx, 2, 16));
      mx = fmaxf(mx, __shfl_xor(mx, 4, 16));
      mx = fmaxf(mx, __shfl_xor(mx, 8, 16));
      float alpha = __expf(mrow[j] - mx);
      mrow[j] = mx;
      float rs = 0.f;
      #pragma unroll
      for (int sf = 0; sf < SB / 16; sf++) {
        float p = __expf(sc[sf][j] - mx);
        sc[sf][j] = p;
        rs += p;
      }
      rs += __shfl_xor(rs, 1, 16);
      rs += __shfl_xor(rs, 2, 16);
      rs += __shfl_xor(rs, 4, 16);
      rs += __shfl_xor(rs, 8, 16);
      lrow[j] = lrow[j] * alpha + rs;
      #pragma unroll
      for (int f = 0; f < 8; f++) o[f][j] *= alpha;
      // spill P (f16) to LDS in A-frag-ready [q][s] layout (own rows only)
      int rloc = wave * 16 + ((lane & 16) ? 8 + j : j);
      #pragma unroll
      for (int sf = 0; sf < SB / 16; sf++)
        Ps[rloc][sf * 16 + lan] = (_Float16)sc[sf][j];
    }
    // ---- P V ----
    Frag16 pa[SB / 32];
    #pragma unroll
    for (int kk = 0; kk < SB / 32; kk++) {
      pa[kk].h[0] = *(const half8*)&Ps[wave * 16 + lan][kk * 32 + khalf * 8];
      pa[kk].h[1] = *(const half8*)&Ps[wave * 16 + lan][kk * 32 + 16 + khalf * 8];
    }
    #pragma unroll
    for (int hf = 0; hf < 8; hf++) {
      #pragma unroll
      for (int kk = 0; kk < SB / 32; kk++) {
        Frag16 bf;
        bf.h[0] = *(const half8*)&Vts[hf * 16 + lan][kk * 32 + khalf * 8];
        bf.h[1] = *(const half8*)&Vts[hf * 16 + lan][kk * 32 + 16 + khalf * 8];
        o[hf] = wmma_f16(pa[kk], bf, o[hf]);
      }
    }
  }

  // epilogue: normalize and store context (f16)
  #pragma unroll
  for (int j = 0; j < 8; j++) {
    float inv = 1.0f / lrow[j];
    int row = q0 + wave * 16 + ((lane & 16) ? 8 + j : j);
    #pragma unroll
    for (int hf = 0; hf < 8; hf++)
      ctx[(size_t)row * DMODEL + head * HDIM + hf * 16 + lan] =
          (_Float16)(o[hf][j] * inv);
  }
}

// ---------------------------------------------------------------------------
// Host launch
// ---------------------------------------------------------------------------
extern "C" void kernel_launch(void* const* d_in, const int* in_sizes, int n_in,
                              void* d_out, int out_size, void* d_ws, size_t ws_size,
                              hipStream_t stream) {
  (void)in_sizes; (void)n_in; (void)out_size; (void)ws_size;
  const float* x    = (const float*)d_in[0];
  // d_in[1] = attn_mask (causal, implicit in kernel)
  const float* sinp = (const float*)d_in[2];
  const float* cosp = (const float*)d_in[3];
  const float* wq   = (const float*)d_in[4];
  const float* wk   = (const float*)d_in[5];
  const float* wv   = (const float*)d_in[6];
  const float* wo   = (const float*)d_in[7];
  const float* qb   = (const float*)d_in[8];
  const float* kb   = (const float*)d_in[9];
  const float* vb   = (const float*)d_in[10];
  float* out = (float*)d_out;

  char* ws = (char*)d_ws;
  size_t off = 0;
  auto alloc = [&](size_t bytes) {
    void* p = ws + off;
    off += (bytes + 255) & ~(size_t)255;
    return p;
  };
  _Float16* wqkvt = (_Float16*)alloc((size_t)WQKV * DMODEL * 2);     // Bt for QKV
  _Float16* wot   = (_Float16*)alloc((size_t)DMODEL * DMODEL * 2);   // Bt for O
  _Float16* xh    = (_Float16*)alloc((size_t)TSEQ * DMODEL * 2);     // f16 x
  _Float16* qkv   = (_Float16*)alloc((size_t)TSEQ * WQKV * 2);
  _Float16* vtb   = (_Float16*)alloc((size_t)KVHEADS * HDIM * TSEQ * 2);
  _Float16* ctx   = (_Float16*)alloc((size_t)TSEQ * DMODEL * 2);
  float*    bias  = (float*)alloc((size_t)WQKV * 4);

  dim3 tb(32, 8);
  transpose_cvt_kernel<<<dim3(112, 112), tb, 0, stream>>>(wq, wqkvt, DMODEL, DMODEL);
  transpose_cvt_kernel<<<dim3(16, 112),  tb, 0, stream>>>(
      wk, wqkvt + (size_t)KCOL * DMODEL, DMODEL, 512);
  transpose_cvt_kernel<<<dim3(16, 112),  tb, 0, stream>>>(
      wv, wqkvt + (size_t)VCOL * DMODEL, DMODEL, 512);
  transpose_cvt_kernel<<<dim3(112, 112), tb, 0, stream>>>(wo, wot, DMODEL, DMODEL);
  cvt_f32_f16_kernel<<<(TSEQ * DMODEL / 8) / 256, 256, 0, stream>>>(
      x, xh, TSEQ * DMODEL / 8);

  (void)hipMemcpyAsync(bias,        qb, (size_t)3584 * 4, hipMemcpyDeviceToDevice, stream);
  (void)hipMemcpyAsync(bias + 3584, kb, (size_t)512 * 4,  hipMemcpyDeviceToDevice, stream);
  (void)hipMemcpyAsync(bias + 4096, vb, (size_t)512 * 4,  hipMemcpyDeviceToDevice, stream);

  gemm_wmma_kernel<true, _Float16>
      <<<dim3(WQKV / 128, TSEQ / 128), 256, 0, stream>>>(
          xh, wqkvt, bias, qkv, TSEQ, DMODEL, WQKV);

  rope_kernel<<<(TSEQ * 32 * 64) / 256, 256, 0, stream>>>(qkv, sinp, cosp);
  vtrans_kernel<<<(KVHEADS * HDIM * TSEQ) / 256, 256, 0, stream>>>(qkv, vtb);

  flash_attn_kernel<<<dim3(TSEQ / 128, NHEADS), 256, 0, stream>>>(qkv, vtb, ctx);

  gemm_wmma_kernel<false, float>
      <<<dim3(DMODEL / 128, TSEQ / 128), 256, 0, stream>>>(
          ctx, wot, nullptr, out, TSEQ, DMODEL, DMODEL);
}
